// JointNet_58832462020719
// MI455X (gfx1250) — matrix operand order repaired and checked by reading, not compiled, not run on hardware
//
#include <hip/hip_runtime.h>
#include <math.h>

static constexpr int BB = 4;
static constexpr int TT = 256;
static constexpr int UU = 100;
static constexpr int KK = 512;     // INNER
static constexpr int VV = 1024;    // VOCAB

typedef __attribute__((ext_vector_type(16))) __bf16 bf16x16;
typedef __attribute__((ext_vector_type(8)))  __bf16 bf16x8;
typedef __attribute__((ext_vector_type(8)))  float  f32x8;

__device__ __forceinline__ unsigned short f32_to_bf16(float f) {
    unsigned u = __builtin_bit_cast(unsigned, f);
    u += 0x7FFFu + ((u >> 16) & 1u);   // round-to-nearest-even
    return (unsigned short)(u >> 16);
}

// Hardware transcendental tanh (single TRANS op on gfx1250).
__device__ __forceinline__ float fast_tanh(float x) {
#if __has_builtin(__builtin_amdgcn_tanhf)
    return __builtin_amdgcn_tanhf(x);
#else
    float y;
    asm volatile("v_tanh_f32 %0, %1\n\tv_nop" : "=v"(y) : "v"(x));
    return y;
#endif
}

// ---------------- Phase 0: W2 (f32, VVxKK) -> bf16 in workspace ----------------
__global__ __launch_bounds__(256) void w2_to_bf16(const float* __restrict__ W2,
                                                  unsigned short* __restrict__ W2b) {
    int idx = blockIdx.x * 256 + threadIdx.x;      // grid sized to VV*KK exactly
    W2b[idx] = f32_to_bf16(W2[idx]);
}

// ---------------- Phase 1: pe / pd projections (f32) ----------------
// outP[r, i] = dot(X[r, 0:512], W1[i, colOff : colOff+512]); W1 row stride 1024
__global__ __launch_bounds__(256) void proj512(const float* __restrict__ X,
                                               const float* __restrict__ W1,
                                               float* __restrict__ outP,
                                               int colOff) {
    __shared__ float Xs[8][512];
    const int r0 = blockIdx.y * 8;
    const int i  = blockIdx.x * 256 + threadIdx.x;   // 0..511
    for (int idx = threadIdx.x; idx < 8 * 512; idx += 256)
        Xs[idx >> 9][idx & 511] = X[(size_t)(r0 + (idx >> 9)) * 512 + (idx & 511)];
    __syncthreads();
    const float* w = W1 + (size_t)i * 1024 + colOff;
    float acc[8];
#pragma unroll
    for (int j = 0; j < 8; ++j) acc[j] = 0.f;
    for (int k = 0; k < 512; ++k) {
        float wv = w[k];
#pragma unroll
        for (int j = 0; j < 8; ++j) acc[j] = fmaf(Xs[j][k], wv, acc[j]);
    }
#pragma unroll
    for (int j = 0; j < 8; ++j) outP[(size_t)(r0 + j) * 512 + i] = acc[j];
}

// ---------------- Phase 2: fused tanh + bf16 WMMA joint GEMM ----------------
// Block tile: M = 128 rows (32 t x 4 u), N = 128 vocab cols, K chunked by 32.
// Per wave: 2(M) x 4(N) WMMA register block; all 4 B fragments preloaded into
// distinct registers so the 8 global_load_b128 clause and drain with staggered
// loadcnt waits instead of serializing on one register octet.
static constexpr int TILE_T = 32;
static constexpr int TILE_U = 4;
static constexpr int TM     = 128;
static constexpr int LDSROW = 40;   // ushorts per hidden row: 32 data + 8 pad (80 B, conflict-free)

__global__ __launch_bounds__(256) void joint_wmma(
    const float* __restrict__ pe,          // (BB*TT, 512) f32
    const float* __restrict__ pd,          // (BB*UU, 512) f32
    const float* __restrict__ b1,          // (512) f32
    const unsigned short* __restrict__ W2b,// (VV, 512) bf16 bits
    const float* __restrict__ b2,          // (VV) f32
    float* __restrict__ out)               // (BB,TT,UU,VV) f32
{
    __shared__ __align__(16) unsigned short hbuf[2 * TM * LDSROW];

    const int tid  = threadIdx.x;
    const int lane = tid & 31;
    const int wave = tid >> 5;
    const int aw   = wave & 3;    // M group: rows [aw*32, aw*32+32)
    const int nw   = wave >> 2;   // N group: cols [nw*64, nw*64+64)

    const int vbase = blockIdx.x * 128;
    const int t0 = (blockIdx.y / 25) * TILE_T;
    const int u0 = (blockIdx.y % 25) * TILE_U;
    const int b  = blockIdx.z;

    // hidden-producer indices: each wave writes one m-row group, k = lane
    const int kk = tid & 31;
    const int mg = tid >> 5;

    // A-fragment LDS offsets (16-bit A 16x32 layout, ISA 7.12.2):
    // lanes 0-15: row m, K 0..7 (v0-3) and K 16..23 (v4-7)
    // lanes 16-31: row m, K 8..15 and K 24..31
    const int apart = lane >> 4;
    const int arow0 = aw * 32 + (lane & 15);            // mi = 0
    const int aoffA = arow0 * LDSROW + apart * 8;       // mi = 0 frag
    const int aoffB = aoffA + 16 * LDSROW;              // mi = 1 frag (rows +16)

    // B-fragment (32x16): lane n<16 -> col n, K 0..7 and 16..23; lane n+16 -> col n, K 8..15 and 24..31.
    // Column n of B == W2 row, contiguous over K.
    const int bcol   = lane & 15;
    const int bkhalf = (lane >> 4) * 8;
    const int vwave  = vbase + nw * 64;                 // this wave's N origin

    // Per-lane B base pointer: row (vwave + n*16 + bcol), K offset bkhalf.
    const unsigned short* bBase =
        W2b + (size_t)(vwave + bcol) * 512 + bkhalf;

    f32x8 acc[2][4];
#pragma unroll
    for (int mi = 0; mi < 2; ++mi)
#pragma unroll
        for (int n = 0; n < 4; ++n)
#pragma unroll
            for (int j = 0; j < 8; ++j) acc[mi][n][j] = 0.f;

    // Producer bases: pd row (m & 3) == (mg & 3) is loop-invariant; pe row
    // (m >> 2) == (mg >> 2) + 2*i  ->  strided pointer, stride 1024 floats.
    const float* peT = pe + (size_t)(b * TT + t0 + (mg >> 2)) * 512 + kk;
    const float* pdT = pd + (size_t)(b * UU + u0 + (mg & 3)) * 512 + kk;
    const int    hst = mg * LDSROW + kk;                 // LDS store base (ushorts)

    for (int kc = 0; kc < 512; kc += 32) {
        const int bs = ((kc >> 5) & 1) * (TM * LDSROW);   // double buffer

        // Produce hidden chunk: tanh(pe + pd + b1) -> bf16 in LDS (128 x 32)
        const float bias1 = b1[kc + kk] + pdT[kc];        // pd hoisted out of i-loop
#pragma unroll
        for (int i = 0; i < 16; ++i) {
            const float x = peT[kc + (size_t)i * 1024] + bias1;
            hbuf[bs + hst + i * (8 * LDSROW)] = f32_to_bf16(fast_tanh(x));
        }
        __syncthreads();   // single barrier/chunk is safe with double buffering

        // Preload all 4 B fragments (8x global_load_b128, distinct registers)
        bf16x8 blo[4], bhi[4];
#pragma unroll
        for (int n = 0; n < 4; ++n) {
            const unsigned short* bp = bBase + (size_t)(n * 16) * 512 + kc;
            blo[n] = *reinterpret_cast<const bf16x8*>(bp);
            bhi[n] = *reinterpret_cast<const bf16x8*>(bp + 16);
        }

        // Two A fragments from LDS (4x ds_load_b128, conflict-free via 80B stride)
        bf16x16 A[2];
#pragma unroll
        for (int mi = 0; mi < 2; ++mi) {
            const int ao = bs + (mi == 0 ? aoffA : aoffB);
            const bf16x8 alo = *reinterpret_cast<const bf16x8*>(&hbuf[ao]);
            const bf16x8 ahi = *reinterpret_cast<const bf16x8*>(&hbuf[ao + 16]);
            A[mi] = __builtin_shufflevector(alo, ahi,
                0, 1, 2, 3, 4, 5, 6, 7, 8, 9, 10, 11, 12, 13, 14, 15);
        }

#pragma unroll
        for (int n = 0; n < 4; ++n) {
            const bf16x16 Bf = __builtin_shufflevector(blo[n], bhi[n],
                0, 1, 2, 3, 4, 5, 6, 7, 8, 9, 10, 11, 12, 13, 14, 15);
            acc[0][n] = __builtin_amdgcn_wmma_f32_16x16x32_bf16(
                false, A[0], false, Bf, (short)0, acc[0][n], false, false);
            acc[1][n] = __builtin_amdgcn_wmma_f32_16x16x32_bf16(
                false, A[1], false, Bf, (short)0, acc[1][n], false, false);
        }
    }

    // Epilogue: + b2, store. C/D layout: VGPR r, lanes 0-15 -> M=r, lanes 16-31 -> M=r+8.
    const int half = lane >> 4;
#pragma unroll
    for (int mi = 0; mi < 2; ++mi) {
#pragma unroll
        for (int n = 0; n < 4; ++n) {
            const int v = vwave + n * 16 + bcol;
            const float bias = b2[v];
#pragma unroll
            for (int r = 0; r < 8; ++r) {
                const int m = aw * 32 + mi * 16 + half * 8 + r;
                const int t = t0 + (m >> 2);
                const int u = u0 + (m & 3);
                out[(((size_t)b * TT + t) * UU + u) * VV + v] = acc[mi][n][r] + bias;
            }
        }
    }
}

extern "C" void kernel_launch(void* const* d_in, const int* in_sizes, int n_in,
                              void* d_out, int out_size, void* d_ws, size_t ws_size,
                              hipStream_t stream) {
    const float* enc = (const float*)d_in[0];   // (4,256,512)
    const float* dec = (const float*)d_in[1];   // (4,100,512)
    const float* W1  = (const float*)d_in[2];   // (512,1024)
    const float* b1  = (const float*)d_in[3];   // (512)
    const float* W2  = (const float*)d_in[4];   // (1024,512)
    const float* b2  = (const float*)d_in[5];   // (1024)
    float* out = (float*)d_out;

    char* ws = (char*)d_ws;
    unsigned short* W2b = (unsigned short*)ws;                    // 1 MiB
    float* pe = (float*)(ws + (1u << 20));                        // 2 MiB
    float* pd = (float*)(ws + (1u << 20) + (2u << 20));           // 800 KiB

    w2_to_bf16<<<dim3((VV * KK) / 256), 256, 0, stream>>>(W2, W2b);
    proj512<<<dim3(2, (BB * TT) / 8), 256, 0, stream>>>(enc, W1, pe, 0);
    proj512<<<dim3(2, (BB * UU) / 8), 256, 0, stream>>>(dec, W1, pd, 512);
    joint_wmma<<<dim3(VV / 128, (TT / TILE_T) * (UU / TILE_U), BB), 256, 0, stream>>>(
        pe, pd, b1, W2b, b2, out);
}
